// Attention_68719476958
// MI455X (gfx1250) — matrix operand order repaired and checked by reading, not compiled, not run on hardware
//
#include <hip/hip_runtime.h>

// ---------------------------------------------------------------------------
// Fused causal attention for MI455X (gfx1250, wave32, WMMA bf16 16x16x32).
//   pass 0  : convert the four 768x768 weight matrices f32 -> bf16
//   pass 1-3: Q/K/V projections (fp32 act x bf16 W, WMMA; V stored transposed)
//   pass 4  : flash attention   (bf16 QKV, fp32 accum, online softmax,
//                                async global->LDS staging of K / V^T tiles)
//   pass 5  : output projection (bf16 in -> fp32 out, WMMA)
// ---------------------------------------------------------------------------

namespace {
constexpr int BATCH  = 4;
constexpr int SEQ    = 4096;
constexpr int DIM    = 768;
constexpr int MROWS  = BATCH * SEQ;     // 16384
constexpr int BR     = 32;              // query rows per workgroup
constexpr int KC     = 64;              // keys per block
constexpr int DCHUNK = 192;             // D columns per wave in PV
constexpr int NT     = DCHUNK / 16;     // 12 accumulator tiles per wave
constexpr float QK_SCALE = 0.036084391824351615f;  // 1/sqrt(768)
constexpr float NEG_MASK = -1.0e9f;                // matches reference
}

typedef __attribute__((ext_vector_type(16))) __bf16 v16bf;
typedef __attribute__((ext_vector_type(8)))  __bf16 v8bf;
typedef __attribute__((ext_vector_type(8)))  float  v8f;

#define USE_ASYNC_LDS 1

// 16-byte global -> LDS copy. Async variant uses the CDNA5 ASYNCcnt path
// (GLOBAL_LOAD_ASYNC_TO_LDS_B128: VDST = LDS byte address, VADDR = 64b addr).
__device__ __forceinline__ void copy16_g2l(const void* g, void* l) {
#if USE_ASYNC_LDS
  asm volatile("global_load_async_to_lds_b128 %0, %1, off"
               :: "v"((unsigned)(uintptr_t)l),
                  "v"((unsigned long long)(uintptr_t)g)
               : "memory");
#else
  *(uint4*)l = *(const uint4*)g;
#endif
}

__device__ __forceinline__ void wait_async_copies() {
#if USE_ASYNC_LDS
#if __has_builtin(__builtin_amdgcn_s_wait_asynccnt)
  __builtin_amdgcn_s_wait_asynccnt(0);
#else
  asm volatile("s_wait_asynccnt 0x0" ::: "memory");
#endif
#endif
}

__device__ __forceinline__ v8f wmma_bf16(v16bf a, v16bf b, v8f c) {
  return __builtin_amdgcn_wmma_f32_16x16x32_bf16(
      /*neg_a=*/false, a, /*neg_b=*/false, b,
      /*c_mod=*/(short)0, c, /*reuse_a=*/false, /*reuse_b=*/false);
}

// A operand (16x32, MxK): lane = M row; halves j<8 -> k0+8*hi+j, j>=8 -> +16.
__device__ __forceinline__ v16bf load_a_bf16(const __bf16* base, int ld,
                                             int row0, int k0, int lane) {
  const int m  = row0 + (lane & 15);
  const int kb = k0 + ((lane >> 4) << 3);
  const __bf16* p = base + (size_t)m * ld + kb;
  v8bf lo = *(const v8bf*)p;
  v8bf hi = *(const v8bf*)(p + 16);
  v16bf r;
#pragma unroll
  for (int j = 0; j < 8; ++j) { r[j] = lo[j]; r[j + 8] = hi[j]; }
  return r;
}

__device__ __forceinline__ v16bf load_a_f32(const float* base, int ld,
                                            int row0, int k0, int lane) {
  const int m  = row0 + (lane & 15);
  const int kb = k0 + ((lane >> 4) << 3);
  const float* p = base + (size_t)m * ld + kb;
  v16bf r;
#pragma unroll
  for (int j = 0; j < 8; ++j) { r[j] = (__bf16)p[j]; r[j + 8] = (__bf16)p[16 + j]; }
  return r;
}

// B operand (32x16, KxN): lane = N column; halves j<8 -> k0+8*hi+j, j>=8 -> +16.
// Source must have column n contiguous in k (i.e. row-major [N][K]).
__device__ __forceinline__ v16bf load_b_bf16(const __bf16* base, int ld,
                                             int n0, int k0, int lane) {
  const int n  = n0 + (lane & 15);
  const int kb = k0 + ((lane >> 4) << 3);
  const __bf16* p = base + (size_t)n * ld + kb;
  v8bf lo = *(const v8bf*)p;
  v8bf hi = *(const v8bf*)(p + 16);
  v16bf r;
#pragma unroll
  for (int j = 0; j < 8; ++j) { r[j] = lo[j]; r[j + 8] = hi[j]; }
  return r;
}

// ---------------------------------------------------------------------------
// Weight pre-conversion: f32 -> bf16, 8 elements per thread.
// ---------------------------------------------------------------------------
__global__ void __launch_bounds__(256)
f32_to_bf16_kernel(const float* __restrict__ src, __bf16* __restrict__ dst, int n) {
  const int i = (blockIdx.x * 256 + threadIdx.x) * 8;
  if (i + 8 <= n) {
    float4 a = *(const float4*)(src + i);
    float4 b = *(const float4*)(src + i + 4);
    v8bf p;
    p[0] = (__bf16)a.x; p[1] = (__bf16)a.y; p[2] = (__bf16)a.z; p[3] = (__bf16)a.w;
    p[4] = (__bf16)b.x; p[5] = (__bf16)b.y; p[6] = (__bf16)b.z; p[7] = (__bf16)b.w;
    *(v8bf*)(dst + i) = p;
  }
}

// ---------------------------------------------------------------------------
// GEMM: C[M x 768] = A[M x 768] * W[768 x 768]^T + bias   (torch Linear)
// Each wave: 16 rows x 96 cols (6 WMMA tiles), K-loop 24 steps of 32.
// OUT_MODE: 0 = f32 row-major, 1 = bf16 row-major, 2 = bf16 [b][col][s].
// ---------------------------------------------------------------------------
template <bool A_BF16, int OUT_MODE>
__global__ void __launch_bounds__(256)
gemm768(const void* __restrict__ Asrc, const __bf16* __restrict__ W,
        const float* __restrict__ bias, void* __restrict__ Cdst) {
  const int lane = threadIdx.x & 31;
  const int wave = threadIdx.x >> 5;
  const int gw   = blockIdx.x * 8 + wave;
  const int row0 = (gw >> 3) * 16;   // 8 column chunks of 96 per row band
  const int col0 = (gw & 7) * 96;

  const v8f vzero = {0.f, 0.f, 0.f, 0.f, 0.f, 0.f, 0.f, 0.f};
  v8f acc[6];
#pragma unroll
  for (int t = 0; t < 6; ++t) acc[t] = vzero;

  for (int k0 = 0; k0 < DIM; k0 += 32) {
    if (k0 + 32 < DIM) {
      if (A_BF16) __builtin_prefetch((const __bf16*)Asrc + (size_t)row0 * DIM + k0 + 32, 0, 1);
      else        __builtin_prefetch((const float*)Asrc + (size_t)row0 * DIM + k0 + 32, 0, 1);
    }
    v16bf a;
    if (A_BF16) a = load_a_bf16((const __bf16*)Asrc, DIM, row0, k0, lane);
    else        a = load_a_f32((const float*)Asrc, DIM, row0, k0, lane);
#pragma unroll
    for (int t = 0; t < 6; ++t) {
      v16bf b = load_b_bf16(W, DIM, col0 + t * 16, k0, lane);
      acc[t] = wmma_bf16(a, b, acc[t]);
    }
  }

  const int n = lane & 15, hi = lane >> 4;
  if (OUT_MODE == 2) {
    // Transposed bf16 store: [b][col][s]; the 8 D-tile rows are consecutive s.
    const int bb    = row0 / SEQ;
    const int sbase = (row0 % SEQ) + 8 * hi;
#pragma unroll
    for (int t = 0; t < 6; ++t) {
      const int col = col0 + t * 16 + n;
      const float bv = bias[col];
      v8bf pack;
#pragma unroll
      for (int r = 0; r < 8; ++r) pack[r] = (__bf16)(acc[t][r] + bv);
      *(v8bf*)((__bf16*)Cdst + ((size_t)bb * DIM + col) * SEQ + sbase) = pack;
    }
  } else {
#pragma unroll
    for (int t = 0; t < 6; ++t) {
      const int col = col0 + t * 16 + n;
      const float bv = bias[col];
#pragma unroll
      for (int r = 0; r < 8; ++r) {
        const int row = row0 + r + 8 * hi;
        const float v = acc[t][r] + bv;
        if (OUT_MODE == 1) ((__bf16*)Cdst)[(size_t)row * DIM + col] = (__bf16)v;
        else               ((float*)Cdst)[(size_t)row * DIM + col] = v;
      }
    }
  }
}

// ---------------------------------------------------------------------------
// Flash attention, causal. One workgroup = 32 query rows of one batch.
// 8 waves = 2 row groups (16 rows) x 4 D-chunks (192 cols each).
// LDS (dynamic): 96KB K-tile (reused as V^T tile) + scores + P + softmax state.
// Vt is the per-batch transposed value projection [b][d][s].
// ---------------------------------------------------------------------------
__global__ void __launch_bounds__(256)
flash_attn(const __bf16* __restrict__ Q, const __bf16* __restrict__ K,
           const __bf16* __restrict__ Vt, __bf16* __restrict__ O) {
  extern __shared__ char smem[];
  __bf16* sKV = (__bf16*)smem;                           // [KC][DIM] then [DIM][KC]
  float*  sS  = (float*)(smem + (size_t)KC * DIM * 2);   // [BR][KC] scores
  __bf16* sP  = (__bf16*)((char*)sS + (size_t)BR * KC * 4);  // [BR][KC] probs
  float*  sM  = (float*)((char*)sP + (size_t)BR * KC * 2);   // running max
  float*  sL  = sM + BR;                                  // running sum
  float*  sA  = sL + BR;                                  // per-block alpha

  const int tid  = threadIdx.x;
  const int lane = tid & 31;
  const int wave = tid >> 5;
  const int rg   = wave >> 2;   // row group 0..1
  const int dw   = wave & 3;    // D chunk 0..3
  const int b    = blockIdx.y;
  const int q0   = blockIdx.x * BR;
  const int n    = lane & 15, hi = lane >> 4;

  const __bf16* Qb  = Q  + (size_t)b * SEQ * DIM;
  const __bf16* Kb  = K  + (size_t)b * SEQ * DIM;
  const __bf16* Vtb = Vt + (size_t)b * DIM * SEQ;

  if (tid < BR) { sM[tid] = -3.0e38f; sL[tid] = 0.f; }

  const v8f vzero = {0.f, 0.f, 0.f, 0.f, 0.f, 0.f, 0.f, 0.f};
  v8f acc[NT];
#pragma unroll
  for (int t = 0; t < NT; ++t) acc[t] = vzero;

  const int nkeys = q0 + BR;  // causal: keys 0 .. q0+BR-1
  for (int kk0 = 0; kk0 < nkeys; kk0 += KC) {
    __syncthreads();  // previous PV reads of sKV are done
    // ---- stage K tile [KC][DIM] (async global -> LDS, 16B chunks) ----
    {
      const __bf16* src = Kb + (size_t)kk0 * DIM;
      for (int i = tid * 8; i < KC * DIM; i += 256 * 8)
        copy16_g2l(src + i, sKV + i);
      wait_async_copies();
    }
    __syncthreads();
    // ---- scores: each wave one 16x16 tile of Q·K^T ----
    {
      v8f s = vzero;
#pragma unroll 4
      for (int d0 = 0; d0 < DIM; d0 += 32) {
        v16bf a  = load_a_bf16(Qb, DIM, q0 + rg * 16, d0, lane);
        v16bf kb = load_b_bf16(sKV, DIM, dw * 16, d0, lane);
        s = wmma_bf16(a, kb, s);
      }
#pragma unroll
      for (int r = 0; r < 8; ++r)
        sS[(rg * 16 + r + 8 * hi) * KC + dw * 16 + n] = s[r] * QK_SCALE;
    }
    __syncthreads();
    // ---- re-stage buffer as V^T tile [DIM][KC] (K tile is dead) ----
    for (int i = tid; i < DIM * KC / 8; i += 256) {
      const int d = i >> 3;            // row in V^T
      const int c = (i & 7) << 3;      // 8-half chunk within the row
      copy16_g2l(Vtb + (size_t)d * SEQ + kk0 + c, sKV + d * KC + c);
    }
    // ---- online softmax, one thread per row (overlaps V^T staging) ----
    if (tid < BR) {
      const int r    = tid;
      const int qrow = q0 + r;
      const float m0 = sM[r];
      float bmax = -3.0e38f;
      for (int c = 0; c < KC; ++c) {
        float v = sS[r * KC + c];
        if (kk0 + c > qrow) v = NEG_MASK;   // causal mask, matches reference
        sS[r * KC + c] = v;
        bmax = fmaxf(bmax, v);
      }
      const float mn = fmaxf(m0, bmax);
      const float al = __expf(m0 - mn);
      float sum = 0.f;
      for (int c = 0; c < KC; ++c) {
        const float p = __expf(sS[r * KC + c] - mn);
        sP[r * KC + c] = (__bf16)p;
        sum += p;
      }
      sL[r] = sL[r] * al + sum;
      sM[r] = mn;
      sA[r] = al;
    }
    wait_async_copies();
    __syncthreads();
    // ---- rescale O and accumulate P·V over this wave's 192 columns ----
    {
      float al[8];
#pragma unroll
      for (int r = 0; r < 8; ++r) al[r] = sA[rg * 16 + r + 8 * hi];
#pragma unroll
      for (int t = 0; t < NT; ++t)
#pragma unroll
        for (int r = 0; r < 8; ++r) acc[t][r] = acc[t][r] * al[r];

      v16bf p0 = load_a_bf16(sP, KC, rg * 16, 0, lane);
      v16bf p1 = load_a_bf16(sP, KC, rg * 16, 32, lane);
#pragma unroll
      for (int t = 0; t < NT; ++t) {
        v16bf v0 = load_b_bf16(sKV, KC, dw * DCHUNK + t * 16, 0, lane);
        acc[t] = wmma_bf16(p0, v0, acc[t]);
        v16bf v1 = load_b_bf16(sKV, KC, dw * DCHUNK + t * 16, 32, lane);
        acc[t] = wmma_bf16(p1, v1, acc[t]);
      }
    }
  }

  // ---- finalize: O /= l, write bf16 attended ----
  float inv[8];
#pragma unroll
  for (int r = 0; r < 8; ++r) inv[r] = 1.0f / sL[rg * 16 + r + 8 * hi];
#pragma unroll
  for (int t = 0; t < NT; ++t) {
#pragma unroll
    for (int r = 0; r < 8; ++r) {
      const size_t row = (size_t)b * SEQ + q0 + rg * 16 + r + 8 * hi;
      const int    col = dw * DCHUNK + t * 16 + n;
      O[row * DIM + col] = (__bf16)(acc[t][r] * inv[r]);
    }
  }
}

// ---------------------------------------------------------------------------
extern "C" void kernel_launch(void* const* d_in, const int* in_sizes, int n_in,
                              void* d_out, int out_size, void* d_ws, size_t ws_size,
                              hipStream_t stream) {
  (void)in_sizes; (void)n_in; (void)out_size; (void)ws_size;
  const float* query = (const float*)d_in[0];
  const float* key_  = (const float*)d_in[1];
  const float* value = (const float*)d_in[2];
  const float* Wq = (const float*)d_in[3];
  const float* bq = (const float*)d_in[4];
  const float* Wk = (const float*)d_in[5];
  const float* bk = (const float*)d_in[6];
  const float* Wv = (const float*)d_in[7];
  const float* bv = (const float*)d_in[8];
  const float* Wo = (const float*)d_in[9];
  const float* bo = (const float*)d_in[10];
  // d_in[11] is the tril mask; causal masking is done analytically in-kernel.

  const size_t MD = (size_t)MROWS * DIM;
  const size_t WD = (size_t)DIM * DIM;
  __bf16* Qp  = (__bf16*)d_ws;
  __bf16* Kp  = Qp + MD;
  __bf16* Vpt = Kp + MD;   // transposed per batch: [b][d][s]
  __bf16* Att = Vpt + MD;
  __bf16* Wqb = Att + MD;
  __bf16* Wkb = Wqb + WD;
  __bf16* Wvb = Wkb + WD;
  __bf16* Wob = Wvb + WD;  // total ws: 4*MD*2 + 4*WD*2 ≈ 100.7 MB

  dim3 blk(256);
  const int cvtGrid = (int)(WD / 8 / 256);  // 288
  f32_to_bf16_kernel<<<cvtGrid, blk, 0, stream>>>(Wq, Wqb, (int)WD);
  f32_to_bf16_kernel<<<cvtGrid, blk, 0, stream>>>(Wk, Wkb, (int)WD);
  f32_to_bf16_kernel<<<cvtGrid, blk, 0, stream>>>(Wv, Wvb, (int)WD);
  f32_to_bf16_kernel<<<cvtGrid, blk, 0, stream>>>(Wo, Wob, (int)WD);

  dim3 gemmGrid(MROWS / 16);  // 1024 blocks * 8 waves = one 16x96 tile per wave
  gemm768<false, 1><<<gemmGrid, blk, 0, stream>>>(query, Wqb, bq, Qp);
  gemm768<false, 1><<<gemmGrid, blk, 0, stream>>>(key_,  Wkb, bk, Kp);
  gemm768<false, 2><<<gemmGrid, blk, 0, stream>>>(value, Wvb, bv, Vpt);

  const size_t smem = (size_t)KC * DIM * 2      // K / V^T tile (96 KB)
                    + (size_t)BR * KC * 4       // scores
                    + (size_t)BR * KC * 2       // P (bf16)
                    + 3 * BR * 4;               // m, l, alpha
  flash_attn<<<dim3(SEQ / BR, BATCH), blk, smem, stream>>>(Qp, Kp, Vpt, Att);

  gemm768<true, 0><<<gemmGrid, blk, 0, stream>>>(Att, Wob, bo, (float*)d_out);
}
